// EncoderBlock_39668317946560
// MI455X (gfx1250) — matrix operand order repaired and checked by reading, not compiled, not run on hardware
//
#include <hip/hip_runtime.h>

#define B_     128
#define P_     196
#define C_     3
#define HID_   768
#define HEADS_ 12
#define HD_    64
#define FF_    3072
#define MTOT   (B_ * P_)          // 25088 tokens

typedef _Float16 f16;
typedef __attribute__((ext_vector_type(16))) _Float16 v16h;
typedef __attribute__((ext_vector_type(8)))  _Float16 v8h;
typedef __attribute__((ext_vector_type(8)))  float    v8f;

#if __has_builtin(__builtin_amdgcn_global_load_async_to_lds_b128)
#define USE_ASYNC 1
#else
#define USE_ASYNC 0
#endif

#if USE_ASYNC
#define AS1 __attribute__((address_space(1)))
#define AS3 __attribute__((address_space(3)))
typedef int v4i_ __attribute__((vector_size(4 * sizeof(int))));  // matches builtin param type
static __device__ __forceinline__ void async_b128(const void* g, void* l) {
  // global -> LDS async copy, 16B per lane, tracked by ASYNCcnt
  __builtin_amdgcn_global_load_async_to_lds_b128((AS1 v4i_*)g, (AS3 v4i_*)l, 0, 0);
}
#endif

// ---------------------------------------------------------------- WMMA helper
static __device__ __forceinline__ v8f wmma16(v16h a, v16h b, v8f c) {
  // D = A(16x32 f16) * B(32x16 f16) + C(16x16 f32)
  return __builtin_amdgcn_wmma_f32_16x16x32_f16(false, a, false, b, (short)0, c,
                                                false, false);
}
static __device__ __forceinline__ v16h ld2x8(const f16* p0, const f16* p1) {
  v16h r;
  *(v8h*)&r       = *(const v8h*)p0;   // halves 0..7
  *((v8h*)&r + 1) = *(const v8h*)p1;   // halves 8..15
  return r;
}

// ---------------------------------------------------------------- f32 -> f16
__global__ void cvt_f32_f16(const float* __restrict__ in, f16* __restrict__ out, int n) {
  int i = blockIdx.x * blockDim.x + threadIdx.x;
  if (i < n) out[i] = (f16)in[i];
}

// ---------------------------------------------------------------- highpass
// out = X - Re(A X A^T), A circulant with taps a[d] = (1/16) sum_{f=-4}^{3} cis(2*pi*f*d/16)
__global__ __launch_bounds__(256) void highpass_kernel(const float* __restrict__ x,
                                                       float* __restrict__ hp) {
  __shared__ float arS[16], aiS[16];
  __shared__ float Xs[8][16][17];
  __shared__ float Yr[8][16][17];
  __shared__ float Yi[8][16][17];
  const int tid = threadIdx.x, wave = tid >> 5, lane = tid & 31;
  if (tid < 16) {
    const float TWO_PI = 6.283185307179586f;
    float sr = 0.f, si = 0.f;
    for (int f = -4; f <= 3; ++f) {
      float ang = TWO_PI * (float)f * (float)tid * (1.0f / 16.0f);
      sr += cosf(ang);
      si += sinf(ang);
    }
    arS[tid] = sr * (1.0f / 16.0f);
    aiS[tid] = si * (1.0f / 16.0f);
  }
  const int pc = blockIdx.x * 8 + wave;              // (b*P + p)*3 + c
  const float* src = x + (size_t)pc * 256;
  float xv[8];
#pragma unroll
  for (int j = 0; j < 8; ++j) {
    int e = lane * 8 + j;
    xv[j] = src[e];
    Xs[wave][e >> 4][e & 15] = xv[j];
  }
  __syncthreads();
#pragma unroll
  for (int j = 0; j < 8; ++j) {
    int e = lane * 8 + j, xx = e >> 4, yy = e & 15;
    float yr = 0.f, yi = 0.f;
    for (int u = 0; u < 16; ++u) {
      float xu = Xs[wave][u][yy];
      int d = (xx - u) & 15;
      yr += arS[d] * xu;
      yi += aiS[d] * xu;
    }
    Yr[wave][xx][yy] = yr;
    Yi[wave][xx][yy] = yi;
  }
  __syncthreads();
  float* dst = hp + (size_t)pc * 256;
#pragma unroll
  for (int j = 0; j < 8; ++j) {
    int e = lane * 8 + j, xx = e >> 4, yy = e & 15;
    float zr = 0.f;
    for (int v = 0; v < 16; ++v) {
      int d = (yy - v) & 15;
      zr += Yr[wave][xx][v] * arS[d] - Yi[wave][xx][v] * aiS[d];
    }
    dst[e] = xv[j] - zr;
  }
}

// ---------------------------------------------------------------- layernorm (1 wave / row)
__global__ __launch_bounds__(256) void ln_kernel(const float* __restrict__ src,
                                                 const float* __restrict__ resid,
                                                 const float* __restrict__ g,
                                                 const float* __restrict__ bta,
                                                 float* __restrict__ out32,
                                                 f16* __restrict__ out16) {
  const int row = blockIdx.x * 8 + (threadIdx.x >> 5);
  const int lane = threadIdx.x & 31;
  const float* s = src + (size_t)row * HID_;
  float v[24];
  float sum = 0.f;
#pragma unroll
  for (int i = 0; i < 24; ++i) { v[i] = s[i * 32 + lane]; sum += v[i]; }
#pragma unroll
  for (int off = 1; off < 32; off <<= 1) sum += __shfl_xor(sum, off, 32);
  const float mean = sum * (1.0f / HID_);
  float var = 0.f;
#pragma unroll
  for (int i = 0; i < 24; ++i) { float d = v[i] - mean; var += d * d; }
#pragma unroll
  for (int off = 1; off < 32; off <<= 1) var += __shfl_xor(var, off, 32);
  const float rstd = rsqrtf(var * (1.0f / HID_) + 1e-5f);
#pragma unroll
  for (int i = 0; i < 24; ++i) {
    int c = i * 32 + lane;
    float y = (v[i] - mean) * rstd * g[c] + bta[c];
    if (resid) y += resid[(size_t)row * HID_ + c];
    if (out32) out32[(size_t)row * HID_ + c] = y;
    if (out16) out16[(size_t)row * HID_ + c] = (f16)y;
  }
}

// ---------------------------------------------------------------- WMMA GEMM
// out[M,N] = act(A[M,K] @ W[N,K]^T + bias);  A,W f16 row-major; M%256==0, N%64==0, K%32==0
// 256 threads = 8 waves; workgroup tile 256x64; wave tile 32x64 (8 WMMA / K-step).
#define ASTR 40
#define BSTR 40
__global__ __launch_bounds__(256) void gemm_kernel(const f16* __restrict__ A,
                                                   const f16* __restrict__ W,
                                                   const float* __restrict__ bias,
                                                   float* __restrict__ out32,
                                                   f16* __restrict__ out16,
                                                   int N, int K, int relu) {
#if USE_ASYNC
  __shared__ f16 Ash[2][256 * ASTR];
  __shared__ f16 Bsh[2][64 * BSTR];
#else
  __shared__ f16 Ash[1][256 * ASTR];
  __shared__ f16 Bsh[1][64 * BSTR];
#endif
  const int tid = threadIdx.x, wave = tid >> 5, lane = tid & 31;
  const int l15 = lane & 15, hsel = lane >> 4;
  const int m0 = blockIdx.y * 256;
  const int n0 = blockIdx.x * 64;
  v8f acc[2][4] = {};
  const int ksteps = K >> 5;

#if USE_ASYNC
  // each thread moves 4x16B of A (rows tid>>2 + i*64, 8-half chunk (tid&3)*8) and 1x16B of W
  const int arow = tid >> 2, achk = (tid & 3) * 8;
#pragma unroll
  for (int i = 0; i < 4; ++i) {
    int row = arow + i * 64;
    async_b128(A + (size_t)(m0 + row) * K + achk, &Ash[0][row * ASTR + achk]);
  }
  async_b128(W + (size_t)(n0 + arow) * K + achk, &Bsh[0][arow * BSTR + achk]);
  for (int kt = 0; kt < ksteps; ++kt) {
    const int cur = kt & 1, nxt = cur ^ 1;
    if (kt + 1 < ksteps) {
      const size_t kb = (size_t)(kt + 1) << 5;
#pragma unroll
      for (int i = 0; i < 4; ++i) {
        int row = arow + i * 64;
        async_b128(A + (size_t)(m0 + row) * K + kb + achk, &Ash[nxt][row * ASTR + achk]);
      }
      async_b128(W + (size_t)(n0 + arow) * K + kb + achk, &Bsh[nxt][arow * BSTR + achk]);
      // async loads complete in order: <=5 outstanding => tile kt has landed
      asm volatile("s_wait_asynccnt 0x5" ::: "memory");
    } else {
      asm volatile("s_wait_asynccnt 0x0" ::: "memory");
    }
    __syncthreads();
    {
      const f16* ab = &Ash[cur][0];
      const f16* bb = &Bsh[cur][0];
      const f16* ap0 = ab + (wave * 32 + l15) * ASTR + hsel * 8;
      v16h a0 = ld2x8(ap0, ap0 + 16);
      const f16* ap1 = ap0 + 16 * ASTR;
      v16h a1 = ld2x8(ap1, ap1 + 16);
#pragma unroll
      for (int n = 0; n < 4; ++n) {
        const f16* bp = bb + (n * 16 + l15) * BSTR + hsel * 8;
        v16h b = ld2x8(bp, bp + 16);
        acc[0][n] = wmma16(a0, b, acc[0][n]);
        acc[1][n] = wmma16(a1, b, acc[1][n]);
      }
    }
    __syncthreads();
  }
#else
  for (int kt = 0; kt < ksteps; ++kt) {
    const int kb = kt << 5;
    {  // stage A tile 256x32: one row (32 halfs) per thread
      const f16* sp = A + (size_t)(m0 + tid) * K + kb;
      f16* dp = &Ash[0][tid * ASTR];
      *(v8h*)(dp)      = *(const v8h*)(sp);
      *(v8h*)(dp + 8)  = *(const v8h*)(sp + 8);
      *(v8h*)(dp + 16) = *(const v8h*)(sp + 16);
      *(v8h*)(dp + 24) = *(const v8h*)(sp + 24);
      if (kt + 1 < ksteps) __builtin_prefetch(sp + 32, 0, 1);  // global_prefetch_b8
    }
    if (tid < 64) {  // stage W tile 64x32
      const f16* sp = W + (size_t)(n0 + tid) * K + kb;
      f16* dp = &Bsh[0][tid * BSTR];
      *(v8h*)(dp)      = *(const v8h*)(sp);
      *(v8h*)(dp + 8)  = *(const v8h*)(sp + 8);
      *(v8h*)(dp + 16) = *(const v8h*)(sp + 16);
      *(v8h*)(dp + 24) = *(const v8h*)(sp + 24);
      if (kt + 1 < ksteps) __builtin_prefetch(sp + 32, 0, 1);
    }
    __syncthreads();
    {
      const f16* ap0 = &Ash[0][(wave * 32 + l15) * ASTR + hsel * 8];
      v16h a0 = ld2x8(ap0, ap0 + 16);
      const f16* ap1 = ap0 + 16 * ASTR;
      v16h a1 = ld2x8(ap1, ap1 + 16);
#pragma unroll
      for (int n = 0; n < 4; ++n) {
        const f16* bp = &Bsh[0][(n * 16 + l15) * BSTR + hsel * 8];
        v16h b = ld2x8(bp, bp + 16);
        acc[0][n] = wmma16(a0, b, acc[0][n]);
        acc[1][n] = wmma16(a1, b, acc[1][n]);
      }
    }
    __syncthreads();
  }
#endif

#pragma unroll
  for (int s = 0; s < 2; ++s) {
#pragma unroll
    for (int n = 0; n < 4; ++n) {
      const int col = n0 + n * 16 + l15;
      const float bs = bias ? bias[col] : 0.f;
#pragma unroll
      for (int r = 0; r < 8; ++r) {
        float vv = acc[s][n][r] + bs;
        if (relu) vv = fmaxf(vv, 0.f);
        size_t idx = (size_t)(m0 + wave * 32 + s * 16 + r + 8 * hsel) * N + col;
        if (out32) out32[idx] = vv;
        if (out16) out16[idx] = (f16)vv;
      }
    }
  }
}

// ---------------------------------------------------------------- fused attention
// one block per (batch, head): scores WMMA -> softmax -> ctx WMMA
#define PKP  224      // keys padded to 7*32
#define KSTR 72       // K row stride (halfs)
#define VSTR 232      // V^T row stride
#define PSTR 232      // probs row stride
#define SMEM_ATTN ((PKP * KSTR + 64 * VSTR + 8 * 16 * PSTR) * 2)
__global__ __launch_bounds__(256) void attn_kernel(const f16* __restrict__ q,
                                                   const f16* __restrict__ k,
                                                   const f16* __restrict__ v,
                                                   float* __restrict__ ctx) {
  extern __shared__ char smem_raw[];
  f16* Ksh = (f16*)smem_raw;            // [PKP][KSTR]   K, row = key
  f16* Vts = Ksh + PKP * KSTR;          // [64][VSTR]    V^T, row = head-dim
  f16* Psh = Vts + 64 * VSTR;           // [8][16][PSTR] per-wave probs
  const int bh = blockIdx.x, b = bh / HEADS_, h = bh % HEADS_;
  const int tid = threadIdx.x, wave = tid >> 5, lane = tid & 31;
  const int l15 = lane & 15, hsel = lane >> 4;
  const size_t base = (size_t)b * P_ * HID_ + (size_t)h * HD_;

  // stage V transposed (zero-pad keys 196..223)
  for (int r = tid >> 2; r < PKP; r += 64) {
    const int q4 = tid & 3;
    f16 tv[16];
    if (r < P_) {
      const f16* vs = v + base + (size_t)r * HID_ + q4 * 16;
#pragma unroll
      for (int j = 0; j < 16; ++j) tv[j] = vs[j];
    } else {
#pragma unroll
      for (int j = 0; j < 16; ++j) tv[j] = (f16)0.f;
    }
#pragma unroll
    for (int j = 0; j < 16; ++j) Vts[(q4 * 16 + j) * VSTR + r] = tv[j];
  }
  // stage K row-major
#if USE_ASYNC
  for (int r = tid >> 2; r < P_; r += 64) {
    const int c = (tid & 3) * 16;
    async_b128(k + base + (size_t)r * HID_ + c,     Ksh + r * KSTR + c);
    async_b128(k + base + (size_t)r * HID_ + c + 8, Ksh + r * KSTR + c + 8);
  }
  for (int i = tid; i < 28 * 64; i += 256) {   // zero pad rows 196..223
    int row = P_ + (i >> 6), col = i & 63;
    Ksh[row * KSTR + col] = (f16)0.f;
  }
  asm volatile("s_wait_asynccnt 0x0" ::: "memory");
#else
  for (int r = tid >> 2; r < PKP; r += 64) {
    const int q4 = tid & 3;
    f16 tk[16];
    if (r < P_) {
      const f16* ks = k + base + (size_t)r * HID_ + q4 * 16;
#pragma unroll
      for (int j = 0; j < 16; ++j) tk[j] = ks[j];
    } else {
#pragma unroll
      for (int j = 0; j < 16; ++j) tk[j] = (f16)0.f;
    }
#pragma unroll
    for (int j = 0; j < 16; ++j) Ksh[r * KSTR + q4 * 16 + j] = tk[j];
  }
#endif
  __syncthreads();

  f16* pw = Psh + wave * 16 * PSTR;
  for (int t = wave; t < 13; t += 8) {        // q tiles 0..12 cover rows 0..207
    int row = t * 16 + l15;
    if (row > P_ - 1) row = P_ - 1;           // clamp, results masked on store
    const f16* qp = q + base + (size_t)row * HID_;
    v16h a0 = ld2x8(qp + hsel * 8,      qp + hsel * 8 + 16);       // K 0..31
    v16h a1 = ld2x8(qp + 32 + hsel * 8, qp + 32 + hsel * 8 + 16);  // K 32..63

    v8f s[13];
#pragma unroll
    for (int kt = 0; kt < 13; ++kt) {
      const f16* kp = Ksh + (kt * 16 + l15) * KSTR + hsel * 8;
      v16h b0 = ld2x8(kp, kp + 16);           // k 0..31
      v16h b1 = ld2x8(kp + 32, kp + 48);      // k 32..63
      v8f c = {};
      c = wmma16(a0, b0, c);
      c = wmma16(a1, b1, c);
      s[kt] = c * 0.125f;                     // 1/sqrt(64)
    }
    if (l15 >= 4) {                           // keys 196..207 invalid
#pragma unroll
      for (int r = 0; r < 8; ++r) s[12][r] = -1e30f;
    }

#pragma unroll
    for (int r = 0; r < 8; ++r) {
      float m = -1e30f;
#pragma unroll
      for (int kt = 0; kt < 13; ++kt) m = fmaxf(m, s[kt][r]);
#pragma unroll
      for (int off = 1; off < 16; off <<= 1) m = fmaxf(m, __shfl_xor(m, off, 16));
      float pr[13], sum = 0.f;
#pragma unroll
      for (int kt = 0; kt < 13; ++kt) { float e = __expf(s[kt][r] - m); pr[kt] = e; sum += e; }
#pragma unroll
      for (int off = 1; off < 16; off <<= 1) sum += __shfl_xor(sum, off, 16);
      const float inv = 1.0f / sum;
      const int prow = (r + 8 * hsel) * PSTR;
#pragma unroll
      for (int kt = 0; kt < 13; ++kt) pw[prow + kt * 16 + l15] = (f16)(pr[kt] * inv);
      pw[prow + 208 + l15] = (f16)0.f;        // zero pad tile 13
    }
    asm volatile("s_wait_dscnt 0" ::: "memory");  // probs LDS RAW (same wave, DS in-order)

    v8f acc[4] = {};
#pragma unroll
    for (int kc = 0; kc < 7; ++kc) {
      const f16* pp = pw + l15 * PSTR + kc * 32 + hsel * 8;
      v16h ap = ld2x8(pp, pp + 16);
#pragma unroll
      for (int n = 0; n < 4; ++n) {
        const f16* vp = Vts + (n * 16 + l15) * VSTR + kc * 32 + hsel * 8;
        v16h bp = ld2x8(vp, vp + 16);
        acc[n] = wmma16(ap, bp, acc[n]);
      }
    }
#pragma unroll
    for (int n = 0; n < 4; ++n) {
#pragma unroll
      for (int r = 0; r < 8; ++r) {
        int grow = t * 16 + r + 8 * hsel;
        if (grow < P_)
          ctx[base + (size_t)grow * HID_ + n * 16 + l15] = acc[n][r];
      }
    }
  }
}

// ---------------------------------------------------------------- launcher
extern "C" void kernel_launch(void* const* d_in, const int* in_sizes, int n_in,
                              void* d_out, int out_size, void* d_ws, size_t ws_size,
                              hipStream_t stream) {
  const float* x    = (const float*)d_in[0];
  const float* q_w  = (const float*)d_in[1];
  const float* q_b  = (const float*)d_in[2];
  const float* k_w  = (const float*)d_in[3];
  const float* k_b  = (const float*)d_in[4];
  const float* v_w  = (const float*)d_in[5];
  const float* v_b  = (const float*)d_in[6];
  const float* w1   = (const float*)d_in[7];
  const float* b1   = (const float*)d_in[8];
  const float* w2   = (const float*)d_in[9];
  const float* b2   = (const float*)d_in[10];
  const float* ln0g = (const float*)d_in[11];
  const float* ln0b = (const float*)d_in[12];
  const float* ln1g = (const float*)d_in[13];
  const float* ln1b = (const float*)d_in[14];
  const float* ln2g = (const float*)d_in[15];
  const float* ln2b = (const float*)d_in[16];
  float* out = (float*)d_out;

  char* wsp = (char*)d_ws;
  auto carve = [&](size_t bytes) -> void* {
    void* p = (void*)wsp;
    wsp += (bytes + 255) & ~(size_t)255;
    return p;
  };
  float* hp   = (float*)carve((size_t)MTOT * HID_ * 4);
  float* ctx  = (float*)carve((size_t)MTOT * HID_ * 4);   // reused for MLP f32 out
  float* h2   = (float*)carve((size_t)MTOT * HID_ * 4);
  f16* h16    = (f16*)carve((size_t)MTOT * HID_ * 2);
  f16* qh     = (f16*)carve((size_t)MTOT * HID_ * 2);
  f16* kh     = (f16*)carve((size_t)MTOT * HID_ * 2);
  f16* vh     = (f16*)carve((size_t)MTOT * HID_ * 2);
  f16* h2h    = (f16*)carve((size_t)MTOT * HID_ * 2);
  f16* m1     = (f16*)carve((size_t)MTOT * FF_ * 2);
  f16* qw16   = (f16*)carve((size_t)HID_ * HID_ * 2);
  f16* kw16   = (f16*)carve((size_t)HID_ * HID_ * 2);
  f16* vw16   = (f16*)carve((size_t)HID_ * HID_ * 2);
  f16* w116   = (f16*)carve((size_t)FF_ * HID_ * 2);
  f16* w216   = (f16*)carve((size_t)HID_ * FF_ * 2);

  const int nw = HID_ * HID_, nf = FF_ * HID_;
  cvt_f32_f16<<<(nw + 255) / 256, 256, 0, stream>>>(q_w, qw16, nw);
  cvt_f32_f16<<<(nw + 255) / 256, 256, 0, stream>>>(k_w, kw16, nw);
  cvt_f32_f16<<<(nw + 255) / 256, 256, 0, stream>>>(v_w, vw16, nw);
  cvt_f32_f16<<<(nf + 255) / 256, 256, 0, stream>>>(w1, w116, nf);
  cvt_f32_f16<<<(nf + 255) / 256, 256, 0, stream>>>(w2, w216, nf);

  highpass_kernel<<<(MTOT * C_) / 8, 256, 0, stream>>>(x, hp);

  ln_kernel<<<MTOT / 8, 256, 0, stream>>>(hp, nullptr, ln0g, ln0b, nullptr, h16);

  dim3 gqkv(HID_ / 64, MTOT / 256);
  gemm_kernel<<<gqkv, 256, 0, stream>>>(h16, qw16, q_b, nullptr, qh, HID_, HID_, 0);
  gemm_kernel<<<gqkv, 256, 0, stream>>>(h16, kw16, k_b, nullptr, kh, HID_, HID_, 0);
  gemm_kernel<<<gqkv, 256, 0, stream>>>(h16, vw16, v_b, nullptr, vh, HID_, HID_, 0);

  attn_kernel<<<B_ * HEADS_, 256, SMEM_ATTN, stream>>>(qh, kh, vh, ctx);

  ln_kernel<<<MTOT / 8, 256, 0, stream>>>(ctx, hp, ln1g, ln1b, h2, h2h);

  gemm_kernel<<<dim3(FF_ / 64, MTOT / 256), 256, 0, stream>>>(h2h, w116, b1, nullptr, m1, FF_, HID_, 1);
  gemm_kernel<<<dim3(HID_ / 64, MTOT / 256), 256, 0, stream>>>(m1, w216, b2, ctx, nullptr, HID_, FF_, 0);

  ln_kernel<<<MTOT / 8, 256, 0, stream>>>(ctx, h2, ln2g, ln2b, out, nullptr);

  (void)in_sizes; (void)n_in; (void)out_size; (void)ws_size;
}
